// GATLayer_48868137894059
// MI455X (gfx1250) — compile-verified, matrix operand order checked by reading, not run on hardware
//
#include <hip/hip_runtime.h>
#include <hip/hip_bf16.h>

typedef __attribute__((ext_vector_type(16))) _Float16 v16h;
typedef __attribute__((ext_vector_type(8)))  float    v8f;

#define ELU_ALPHA 0.2f

__device__ __forceinline__ float elu_f(float v) {
    return v > 0.f ? v : ELU_ALPHA * expm1f(v);
}

// float atomic max via monotone int bit ordering
__device__ __forceinline__ void atomicMaxF(float* addr, float val) {
    if (val >= 0.f) atomicMax((int*)addr, __float_as_int(val));
    else            atomicMin((unsigned int*)addr, (unsigned int)__float_as_int(val));
}

// Fragment-layout inversion (ISA 7.12.2, 16-bit A/B 16x32):
//   k-local kl in [0,32): half = (kl>>3)&1, lane = idx + 16*half,
//   j = (kl&7) + 8*(kl>>4)   (j = element index inside the lane's v16h)

// ---------------------------------------------------------------------------
// Kernel 0 (one-shot): repack W (128x128 f32, k-major) into fragment-major
// f16: wfrag[kc][wtile][lane][j], 32KB total. B fragments then load as one
// aligned 32B chunk per lane (global_load_b128 x2, fully coalesced, L2-hot).
// ---------------------------------------------------------------------------
__global__ __launch_bounds__(256) void gat_pack_w(const float* __restrict__ W,
                                                  _Float16* __restrict__ wfrag) {
    int i = blockIdx.x * 256 + threadIdx.x;      // 128*128 elements
    if (i >= 128 * 128) return;
    int k = i >> 7, col = i & 127;
    int kc = k >> 5, kl = k & 31;
    int w = col >> 4, n = col & 15;
    int lane = n + 16 * ((kl >> 3) & 1);
    int j    = (kl & 7) + 8 * (kl >> 4);
    wfrag[((((kc * 8 + w) * 32) + lane) << 4) + j] = (_Float16)W[i];
}

// ---------------------------------------------------------------------------
// Kernel 1: xp = x @ W via f16 WMMA (f32 accum). Block = 256 threads (8 waves),
// owns 32 rows (two M-tiles); wave w owns cols [16w,16w+16). x rows staged in
// LDS in fragment order (8KB) -> fragment loads are ds_load_b128 pairs; B
// fragments come straight from the pre-swizzled global wfrag. Each B fragment
// feeds two WMMAs. Epilogue: one base address + unguarded store clause in the
// interior (only the final block takes the guarded path).
// ---------------------------------------------------------------------------
__global__ __launch_bounds__(256) void gat_gemm_wmma(const float* __restrict__ x,
                                                     const _Float16* __restrict__ wfrag,
                                                     float* __restrict__ xp, int N) {
    __shared__ _Float16 ldsA[2 * 4 * 32 * 16];   // [mtile][kc][lane][j], 8KB

    const int tid  = threadIdx.x;
    const int row0 = blockIdx.x * 32;

    // Stage 32 rows of x as f16 in fragment order (16 elements per thread)
    for (int i = tid; i < 32 * 128; i += 256) {
        int r = i >> 7, c = i & 127;
        int gr = row0 + r;
        float v = (gr < N) ? x[(size_t)gr * 128 + c] : 0.f;
        int mt = r >> 4, rr = r & 15;
        int kc = c >> 5, kl = c & 31;
        int lane = rr + 16 * ((kl >> 3) & 1);
        int j    = (kl & 7) + 8 * (kl >> 4);
        ldsA[((((mt * 4 + kc) * 32) + lane) << 4) + j] = (_Float16)v;
    }
    __syncthreads();

    const int w    = tid >> 5;        // wave id -> 16-col tile of W
    const int lane = tid & 31;
    const v16h* wf = (const v16h*)wfrag;
    const v16h* af = (const v16h*)ldsA;

    v8f acc0 = {}, acc1 = {};
    #pragma unroll
    for (int kc = 0; kc < 4; ++kc) {                 // K = 128, chunks of 32
        v16h b  = wf[(kc * 8 + w) * 32 + lane];      // 32B contiguous per lane
        v16h a0 = af[(0 * 4 + kc) * 32 + lane];
        v16h a1 = af[(1 * 4 + kc) * 32 + lane];
        acc0 = __builtin_amdgcn_wmma_f32_16x16x32_f16(false, a0, false, b,
                                                      (short)0, acc0, false, false);
        acc1 = __builtin_amdgcn_wmma_f32_16x16x32_f16(false, a1, false, b,
                                                      (short)0, acc1, false, false);
    }

    // C/D layout: VGPR r -> M = r + (lane<16?0:8), N = lane%16
    const int moff = (lane < 16) ? 0 : 8;
    const int n    = lane & 15;
    float* base = xp + (size_t)(row0 + moff) * 128 + w * 16 + n;

    if (row0 + 32 <= N) {
        // interior block: single base address, 16 stores with immediate offsets
        #pragma unroll
        for (int r = 0; r < 8; ++r) {
            base[(size_t)r * 128]        = acc0[r];
            base[(size_t)(r + 16) * 128] = acc1[r];
        }
    } else {
        #pragma unroll
        for (int r = 0; r < 8; ++r) {
            if (row0 + moff + r < N)      base[(size_t)r * 128]        = acc0[r];
            if (row0 + moff + r + 16 < N) base[(size_t)(r + 16) * 128] = acc1[r];
        }
    }
}

// ---------------------------------------------------------------------------
// Kernel 2: per-node attention scores. One wave32 per node; lane = feature,
// k = head (F == 32 == wave width). Shuffle-reduce per head.
// ---------------------------------------------------------------------------
__global__ __launch_bounds__(256) void gat_scores(const float* __restrict__ xp,
                                                  const float* __restrict__ a_src,
                                                  const float* __restrict__ a_tgt,
                                                  float* __restrict__ s_src,
                                                  float* __restrict__ s_tgt, int N) {
    const int wid  = (blockIdx.x * blockDim.x + threadIdx.x) >> 5;
    const int lane = threadIdx.x & 31;
    if (wid >= N) return;

    float ps[4], pt[4];
    #pragma unroll
    for (int k = 0; k < 4; ++k) {
        float v = xp[(size_t)wid * 128 + k * 32 + lane];
        ps[k] = v * a_src[k * 32 + lane];
        pt[k] = v * a_tgt[k * 32 + lane];
    }
    #pragma unroll
    for (int off = 16; off > 0; off >>= 1) {
        #pragma unroll
        for (int k = 0; k < 4; ++k) {
            ps[k] += __shfl_xor(ps[k], off, 32);
            pt[k] += __shfl_xor(pt[k], off, 32);
        }
    }
    if (lane == 0) {
        #pragma unroll
        for (int k = 0; k < 4; ++k) {
            s_src[wid * 4 + k] = ps[k];
            s_tgt[wid * 4 + k] = pt[k];
        }
    }
}

// ---------------------------------------------------------------------------
// Kernel 3: init m = -inf, denom = 0, out = 0
// ---------------------------------------------------------------------------
__global__ __launch_bounds__(256) void gat_init(float* __restrict__ m,
                                                float* __restrict__ denom,
                                                float* __restrict__ out, int N) {
    int i = blockIdx.x * blockDim.x + threadIdx.x;
    if (i < N * 4) { m[i] = -__builtin_huge_valf(); denom[i] = 0.f; }
    if (i < N * 128) out[i] = 0.f;
}

// ---------------------------------------------------------------------------
// Kernel 4: segment max of ELU scores. One thread per (edge, head).
// ---------------------------------------------------------------------------
__global__ __launch_bounds__(256) void gat_edge_max(const int* __restrict__ ei,
                                                    const float* __restrict__ s_src,
                                                    const float* __restrict__ s_tgt,
                                                    float* __restrict__ m, int E) {
    int tid = blockIdx.x * blockDim.x + threadIdx.x;
    if (tid >= E * 4) return;
    int e = tid >> 2, h = tid & 3;
    int s = ei[2 * e], t = ei[2 * e + 1];
    float val = elu_f(s_src[s * 4 + h] + s_tgt[t * 4 + h]);
    atomicMaxF(&m[t * 4 + h], val);
}

// ---------------------------------------------------------------------------
// Kernel 5: segment sum of exp(e - m[tgt]) into denom (score recomputed).
// ---------------------------------------------------------------------------
__global__ __launch_bounds__(256) void gat_edge_sum(const int* __restrict__ ei,
                                                    const float* __restrict__ s_src,
                                                    const float* __restrict__ s_tgt,
                                                    const float* __restrict__ m,
                                                    float* __restrict__ denom, int E) {
    int tid = blockIdx.x * blockDim.x + threadIdx.x;
    if (tid >= E * 4) return;
    int e = tid >> 2, h = tid & 3;
    int s = ei[2 * e], t = ei[2 * e + 1];
    float mm = m[t * 4 + h];
    if (!(mm == mm) || mm < -1e30f) mm = 0.f;    // where(isfinite(m), m, 0)
    float ex = expf(elu_f(s_src[s * 4 + h] + s_tgt[t * 4 + h]) - mm);
    atomicAdd(&denom[t * 4 + h], ex);
}

// ---------------------------------------------------------------------------
// Kernel 6: out[tgt] += alpha * xp[src]. One wave32 per edge: lane l handles
// element 32k+l of head k -> coalesced 128-float gather + atomic scatter.
// xp/out are L2-resident (25.6MB each vs 192MB L2).
// ---------------------------------------------------------------------------
__global__ __launch_bounds__(256) void gat_edge_aggregate(const int* __restrict__ ei,
                                                          const float* __restrict__ s_src,
                                                          const float* __restrict__ s_tgt,
                                                          const float* __restrict__ m,
                                                          const float* __restrict__ denom,
                                                          const float* __restrict__ xp,
                                                          float* __restrict__ out, int E) {
    const int widx = (blockIdx.x * blockDim.x + threadIdx.x) >> 5;
    const int lane = threadIdx.x & 31;
    if (widx >= E) return;
    const int s = ei[2 * widx], t = ei[2 * widx + 1];
    #pragma unroll
    for (int k = 0; k < 4; ++k) {
        float mm = m[t * 4 + k];
        if (!(mm == mm) || mm < -1e30f) mm = 0.f;
        float ex = expf(elu_f(s_src[s * 4 + k] + s_tgt[t * 4 + k]) - mm);
        float alpha = ex / denom[t * 4 + k];
        float v = xp[(size_t)s * 128 + k * 32 + lane];
        atomicAdd(&out[(size_t)t * 128 + k * 32 + lane], alpha * v);
    }
}

extern "C" void kernel_launch(void* const* d_in, const int* in_sizes, int n_in,
                              void* d_out, int out_size, void* d_ws, size_t ws_size,
                              hipStream_t stream) {
    const float* x     = (const float*)d_in[0];   // N x 128
    const float* W     = (const float*)d_in[1];   // 128 x 128
    const float* a_src = (const float*)d_in[2];   // 4 x 32
    const float* a_tgt = (const float*)d_in[3];   // 4 x 32
    const int*   ei    = (const int*)d_in[4];     // E x 2

    const int N = in_sizes[0] / 128;
    const int E = in_sizes[4] / 2;

    // workspace layout (floats): xp | s_src | s_tgt | m | denom | wfrag(f16)
    float* ws    = (float*)d_ws;
    float* xp    = ws;
    float* s_src = xp    + (size_t)N * 128;
    float* s_tgt = s_src + (size_t)N * 4;
    float* m     = s_tgt + (size_t)N * 4;
    float* denom = m     + (size_t)N * 4;
    _Float16* wfrag = (_Float16*)(denom + (size_t)N * 4);   // 32KB, 32B-aligned
    float* out   = (float*)d_out;

    gat_pack_w<<<(128 * 128 + 255) / 256, 256, 0, stream>>>(W, wfrag);
    gat_gemm_wmma<<<(N + 31) / 32, 256, 0, stream>>>(x, wfrag, xp, N);
    gat_scores<<<(N + 7) / 8, 256, 0, stream>>>(xp, a_src, a_tgt, s_src, s_tgt, N);
    gat_init<<<(N * 128 + 255) / 256, 256, 0, stream>>>(m, denom, out, N);
    gat_edge_max<<<(E * 4 + 255) / 256, 256, 0, stream>>>(ei, s_src, s_tgt, m, E);
    gat_edge_sum<<<(E * 4 + 255) / 256, 256, 0, stream>>>(ei, s_src, s_tgt, m, denom, E);
    gat_edge_aggregate<<<(E + 7) / 8, 256, 0, stream>>>(ei, s_src, s_tgt, m, denom, xp, out, E);
}
